// MaskMultiheadAttention_4587025072476
// MI455X (gfx1250) — matrix operand order repaired and checked by reading, not compiled, not run on hardware
//
#include <hip/hip_runtime.h>

// ---------------------------------------------------------------------------
// MaskMultiheadAttention for MI455X (gfx1250, wave32, WMMA)
// B=4, N=2048, C=256, H=8, Dh=32
//
// Roofline: attn_vis write = 537MB (write-once), attn_mask read = 67MB
// (stream-once)  ->  ~26us floor at 23.3TB/s. Math ~21 GFLOP ->
// trivially covered by f16 WMMA. Strategy: fuse softmax + vis write + masked
// PV so probabilities never hit HBM; one WG per (b,qtile) looping heads so
// the mask is read exactly once; NT cache hints on the streaming data so L2
// is reserved for the heavily reused f16 Q/K/V (12MB) and weights.
// ---------------------------------------------------------------------------

typedef __attribute__((ext_vector_type(16))) _Float16 v16h;
typedef __attribute__((ext_vector_type(8)))  _Float16 v8h;
typedef __attribute__((ext_vector_type(8)))  float    v8f;
typedef __attribute__((ext_vector_type(4)))  float    v4f;

#define WMMA_F16(A_, B_, C_) \
  __builtin_amdgcn_wmma_f32_16x16x32_f16(false, (A_), false, (B_), (short)0, (C_), false, false)

constexpr int Bb = 4, Nn = 2048, Cc = 256, Hh = 8, Dh = 32;
constexpr float SCALE = 0.17677669529663688f; // 32^-0.5

// Convert two 8-float chunks to a v16h (A/B WMMA operand). 16B-aligned chunks.
static __device__ __forceinline__ v16h cvt_f32x16(const float* p0, const float* p1) {
  v4f a = ((const v4f*)p0)[0], b = ((const v4f*)p0)[1];
  v4f c = ((const v4f*)p1)[0], d = ((const v4f*)p1)[1];
  v16h r;
#pragma unroll
  for (int i = 0; i < 4; i++) {
    r[i] = (_Float16)a[i]; r[4 + i] = (_Float16)b[i];
    r[8 + i] = (_Float16)c[i]; r[12 + i] = (_Float16)d[i];
  }
  return r;
}

// Gather two contiguous 8-half chunks (b128 loads) into a v16h.
static __device__ __forceinline__ v16h ld16h(const _Float16* p0, const _Float16* p1) {
  v8h a = *(const v8h*)p0, b = *(const v8h*)p1;
  v16h r;
#pragma unroll
  for (int i = 0; i < 8; i++) { r[i] = a[i]; r[8 + i] = b[i]; }
  return r;
}

// ---------------------------------------------------------------------------
// K0: invm[b,q] = 1 / max_k attn_mask[b,k,q]   (coalesced column reduction)
// Mask is stream-once -> non-temporal loads keep it out of L2.
// ---------------------------------------------------------------------------
__global__ void __launch_bounds__(256) rowmax_inv_kernel(const float* __restrict__ amask,
                                                         float* __restrict__ invm) {
  int b = blockIdx.x >> 3;
  int q = ((blockIdx.x & 7) << 8) + threadIdx.x;
  const float* p = amask + (size_t)b * Nn * Nn + q;
  float m = 0.f;
  for (int k = 0; k < Nn; k++) m = fmaxf(m, __builtin_nontemporal_load(p + (size_t)k * Nn));
  invm[b * Nn + q] = 1.f / m;
}

// ---------------------------------------------------------------------------
// K1: fused QKV projection. One wave per 16x16 output tile (K-loop of 8 WMMAs).
// q,k stored f16 row-major [B*N, C] (WMMA-A/B friendly); v stored transposed
// f16 [B, C, N] so the PV B-tile is contiguous along key tokens.
// ---------------------------------------------------------------------------
__global__ void __launch_bounds__(256) qkv_kernel(
    const float* __restrict__ X,
    const float* __restrict__ Wq, const float* __restrict__ bq,
    const float* __restrict__ Wk, const float* __restrict__ bk,
    const float* __restrict__ Wv, const float* __restrict__ bv,
    _Float16* __restrict__ qh, _Float16* __restrict__ kh, _Float16* __restrict__ vT) {
  __shared__ _Float16 tbuf[8][16][16];
  int lane = threadIdx.x & 31, wave = threadIdx.x >> 5;
  int gw = blockIdx.x * 8 + wave;        // 24576 waves total
  int proj = gw >> 13;                   // 0=q 1=k 2=v (8192 waves each)
  int rem = gw & 8191;
  int tbase = (rem >> 4) * 16;           // token tile (512 tiles)
  int c0 = (rem & 15) * 16;              // channel tile (16 tiles)
  const float* W    = (proj == 0) ? Wq : ((proj == 1) ? Wk : Wv);
  const float* bias = (proj == 0) ? bq : ((proj == 1) ? bk : bv);

  int l15 = lane & 15;
  int hi = (lane >> 4) & 1;
  int off1 = hi ? 8 : 0;                 // A 16-bit lane layout chunk select
  int koff = hi ? 16 : 0;                // B K-half select
  int m0 = hi ? 8 : 0;                   // C/D row base

  const float* Arow = X + (size_t)(tbase + l15) * Cc;           // A row = token
  const float* Brow = W + (size_t)(c0 + l15) * Cc + koff;       // B col = out channel

  v8f acc = {};
#pragma unroll
  for (int kk = 0; kk < 8; kk++) {
    int K0 = kk * 32;
    v16h Am = cvt_f32x16(Arow + K0 + koff, Arow + K0 + koff + 8);
    v16h Bm = cvt_f32x16(Brow + K0, Brow + K0 + 8);
    acc = WMMA_F16(Am, Bm, acc);
  }
  float bs = bias[c0 + l15];
#pragma unroll
  for (int i = 0; i < 8; i++) acc[i] += bs;

  if (proj < 2) {  // transpose C/D layout -> row-major via LDS, then b128 stores
#pragma unroll
    for (int i = 0; i < 8; i++) tbuf[wave][m0 + i][l15] = (_Float16)acc[i];
  }
  __syncthreads();
  if (proj < 2) {
    _Float16* dst = (proj == 0) ? qh : kh;
    v8h val = *(const v8h*)&tbuf[wave][l15][m0];
    *(v8h*)(dst + (size_t)(tbase + l15) * Cc + c0 + m0) = val;
  } else {         // v: direct transposed store, rows (tokens) are contiguous
    v8h val;
#pragma unroll
    for (int i = 0; i < 8; i++) val[i] = (_Float16)acc[i];
    int bidx = tbase / Nn, nbase = tbase % Nn;
    *(v8h*)(vT + ((size_t)bidx * Cc + c0 + l15) * Nn + nbase + m0) = val;
  }
}

// ---------------------------------------------------------------------------
// K2: fused attention. One WG (256 thr / 8 waves) per (b, qtile of 16 rows),
// looping all 8 heads. Mask tile staged once in LDS (pre-normalized).
// Per head: S=QK^T (WMMA) -> LDS f16; rowmax; exp+rowsum (overwrite with E);
// write attn_vis (NT); PV GEMM with A = E*mask*inv_sum built from LDS;
// cross-wave partial reduction through LDS; write pre-Wo context (f16).
// ---------------------------------------------------------------------------
__global__ void __launch_bounds__(256) attn_kernel(
    const _Float16* __restrict__ qh, const _Float16* __restrict__ kh,
    const _Float16* __restrict__ vT, const float* __restrict__ amask,
    const float* __restrict__ invm, float* __restrict__ avis,
    _Float16* __restrict__ xbuf) {
  __shared__ _Float16 Sb[16 * 2048];   // scores -> E (64KB)
  __shared__ _Float16 Mb[16 * 2048];   // normalized mask, [q][k] (64KB)
  __shared__ float red[256];
  __shared__ float rowmaxs[16];
  __shared__ float rowinvs[16];
  __shared__ float qminvs[16];

  int tid = threadIdx.x, lane = tid & 31, wave = tid >> 5;
  int b = blockIdx.x >> 7;
  int qbase = (blockIdx.x & 127) * 16;
  int l15 = lane & 15;
  int hi = (lane >> 4) & 1;
  int m0 = hi ? 8 : 0;
  int off1 = hi ? 8 : 0;
  int koff = hi ? 16 : 0;

  if (tid < 16) qminvs[tid] = invm[b * Nn + qbase + tid];
  __syncthreads();

  // Stage mask once: Mb[q][k] = amask[b][k][qbase+q] * (1/max_k amask[b][:,q])
  // Stream-once data -> non-temporal loads.
  for (int k = tid; k < Nn; k += 256) {
    const v4f* mp = (const v4f*)(amask + (size_t)b * Nn * Nn + (size_t)k * Nn + qbase);
    v4f x0 = __builtin_nontemporal_load(mp + 0);
    v4f x1 = __builtin_nontemporal_load(mp + 1);
    v4f x2 = __builtin_nontemporal_load(mp + 2);
    v4f x3 = __builtin_nontemporal_load(mp + 3);
    float mv[16];
#pragma unroll
    for (int i = 0; i < 4; i++) {
      mv[i] = x0[i]; mv[4 + i] = x1[i]; mv[8 + i] = x2[i]; mv[12 + i] = x3[i];
    }
#pragma unroll
    for (int q = 0; q < 16; q++) Mb[q * 2048 + k] = (_Float16)(mv[q] * qminvs[q]);
  }
  __syncthreads();

  for (int h = 0; h < Hh; ++h) {
    // A tile: q rows (same for all 8 waves; each covers its own key range)
    const _Float16* qrow = qh + (size_t)(b * Nn + qbase + l15) * Cc + h * 32;
    v16h Aq = ld16h(qrow + koff, qrow + koff + 8);

    // ---- pass A: S = scale * Q K^T, 16 key-tiles per wave ----
    for (int kt = wave * 16; kt < wave * 16 + 16; ++kt) {
      int ktok = kt * 16 + l15;
      const _Float16* krow = kh + (size_t)(b * Nn + ktok) * Cc + h * 32 + koff;
      __builtin_prefetch(krow + 16 * Cc, 0, 0);   // global_prefetch next k-tile
      v16h Bk = ld16h(krow, krow + 8);
      v8f S = {};
      S = WMMA_F16(Aq, Bk, S);
      int col = kt * 16 + l15;
#pragma unroll
      for (int i = 0; i < 8; i++) Sb[(m0 + i) * 2048 + col] = (_Float16)(S[i] * SCALE);
    }
    __syncthreads();

    // ---- rowmax over 2048 keys (lane-staggered to dodge bank conflicts) ----
    {
      int r = tid & 15, chunk = tid >> 4;
      const _Float16* row = Sb + r * 2048 + chunk * 128;
      float mx = -1e30f;
      for (int j = 0; j < 128; j++) {
        int jj = (j + lane * 4) & 127;
        mx = fmaxf(mx, (float)row[jj]);
      }
      red[chunk * 16 + r] = mx;
    }
    __syncthreads();
    if (tid < 16) {
      float mx = -1e30f;
#pragma unroll
      for (int c = 0; c < 16; c++) mx = fmaxf(mx, red[c * 16 + tid]);
      rowmaxs[tid] = mx;
    }
    __syncthreads();

    // ---- exp + rowsum, overwrite Sb with E = exp(S - max) ----
    {
      int r = tid & 15, chunk = tid >> 4;
      _Float16* row = Sb + r * 2048 + chunk * 128;
      float mx = rowmaxs[r], s = 0.f;
      for (int j = 0; j < 128; j++) {
        int jj = (j + lane * 4) & 127;
        float e = __expf((float)row[jj] - mx);
        row[jj] = (_Float16)e;
        s += e;
      }
      red[chunk * 16 + r] = s;
    }
    __syncthreads();
    if (tid < 16) {
      float s = 0.f;
#pragma unroll
      for (int c = 0; c < 16; c++) s += red[c * 16 + tid];
      rowinvs[tid] = 1.f / s;
    }
    __syncthreads();

    // ---- attn_vis = E * inv_sum (pre-mask softmax); write-once -> NT stores,
    //      coalesced 64B segments per half-wave ----
    {
      float rinv[8];
#pragma unroll
      for (int i = 0; i < 8; i++) rinv[i] = rowinvs[m0 + i];
      for (int kt = wave * 16; kt < wave * 16 + 16; ++kt) {
        int col = kt * 16 + l15;
        float* vp = avis + ((size_t)(b * Hh + h) * Nn + qbase + m0) * Nn + col;
#pragma unroll
        for (int i = 0; i < 8; i++) {
          float attn = (float)Sb[(m0 + i) * 2048 + col] * rinv[i];
          __builtin_nontemporal_store(attn, vp + (size_t)i * Nn);
        }
      }
    }

    // ---- pass D: x += (E*mask*inv_sum) @ V over this wave's key range ----
    float rinvm = rowinvs[l15];          // A layout: lane == q row
    v8f acc0 = {}, acc1 = {};
#pragma unroll 2
    for (int kb = 0; kb < 8; kb++) {
      int k0 = wave * 256 + kb * 32;
      const _Float16* ep = Sb + l15 * 2048 + k0 + koff;
      const _Float16* wp = Mb + l15 * 2048 + k0 + koff;
      v8h e0 = *(const v8h*)ep, e1 = *(const v8h*)(ep + 8);
      v8h w0 = *(const v8h*)wp, w1 = *(const v8h*)(wp + 8);
      v16h Ap;
#pragma unroll
      for (int i = 0; i < 8; i++) {
        Ap[i]     = (_Float16)((float)e0[i] * (float)w0[i] * rinvm);
        Ap[8 + i] = (_Float16)((float)e1[i] * (float)w1[i] * rinvm);
      }
      const _Float16* v0p = vT + ((size_t)b * Cc + h * 32 + l15) * Nn + k0 + koff;
      const _Float16* v1p = vT + ((size_t)b * Cc + h * 32 + 16 + l15) * Nn + k0 + koff;
      v16h B0 = ld16h(v0p, v0p + 8);
      v16h B1 = ld16h(v1p, v1p + 8);
      acc0 = WMMA_F16(Ap, B0, acc0);
      acc1 = WMMA_F16(Ap, B1, acc1);
    }
    __syncthreads();                     // all Sb/Mb reads done; reuse Sb as f32 scratch

    float* xp = (float*)Sb;
#pragma unroll
    for (int i = 0; i < 8; i++) {
      xp[wave * 512 + lane * 8 + i] = acc0[i];
      xp[wave * 512 + 256 + lane * 8 + i] = acc1[i];
    }
    __syncthreads();
#pragma unroll
    for (int rep = 0; rep < 2; rep++) {
      int j = tid + rep * 256;
      float s = 0.f;
#pragma unroll
      for (int w2 = 0; w2 < 8; w2++) s += xp[w2 * 512 + j];
      int half = j >> 8;
      int r = j & 255;
      int ln = r >> 3, ii = r & 7;
      int mr = ii + ((ln & 16) ? 8 : 0);
      int d = half * 16 + (ln & 15);
      xbuf[(size_t)(b * Nn + qbase + mr) * Cc + h * 32 + d] = (_Float16)s;
    }
    __syncthreads();                     // before next head overwrites Sb/red
  }
}

// ---------------------------------------------------------------------------
// K3: out = query + xbuf @ Wo^T + bo   (one wave per 16x16 tile, 8 WMMAs)
// ---------------------------------------------------------------------------
__global__ void __launch_bounds__(256) outproj_kernel(
    const _Float16* __restrict__ xbuf, const float* __restrict__ Wo,
    const float* __restrict__ bo, const float* __restrict__ X,
    float* __restrict__ outq) {
  int lane = threadIdx.x & 31, wave = threadIdx.x >> 5;
  int gw = blockIdx.x * 8 + wave;        // 8192 waves
  int tbase = (gw >> 4) * 16, c0 = (gw & 15) * 16;
  int l15 = lane & 15, hi = (lane >> 4) & 1;
  int off1 = hi ? 8 : 0, koff = hi ? 16 : 0, m0 = hi ? 8 : 0;
  const _Float16* Arow = xbuf + (size_t)(tbase + l15) * Cc;
  const float* Brow = Wo + (size_t)(c0 + l15) * Cc + koff;
  v8f acc = {};
#pragma unroll
  for (int kk = 0; kk < 8; kk++) {
    int K0 = kk * 32;
    v16h Am = ld16h(Arow + K0 + koff, Arow + K0 + koff + 8);
    v16h Bm = cvt_f32x16(Brow + K0, Brow + K0 + 8);
    acc = WMMA_F16(Am, Bm, acc);
  }
  float bs = bo[c0 + l15];
#pragma unroll
  for (int i = 0; i < 8; i++) {
    size_t idx = (size_t)(tbase + m0 + i) * Cc + c0 + l15;
    float v = X[idx] + acc[i] + bs;
    __builtin_nontemporal_store(v, outq + idx);   // write-once output
  }
}

// ---------------------------------------------------------------------------
extern "C" void kernel_launch(void* const* d_in, const int* in_sizes, int n_in,
                              void* d_out, int out_size, void* d_ws, size_t ws_size,
                              hipStream_t stream) {
  (void)in_sizes; (void)n_in; (void)out_size; (void)ws_size;
  const float* query = (const float*)d_in[0];
  const float* amask = (const float*)d_in[1];
  const float* Wq = (const float*)d_in[2]; const float* bq = (const float*)d_in[3];
  const float* Wk = (const float*)d_in[4]; const float* bk = (const float*)d_in[5];
  const float* Wv = (const float*)d_in[6]; const float* bv = (const float*)d_in[7];
  const float* Wo = (const float*)d_in[8]; const float* bo = (const float*)d_in[9];

  float* outq = (float*)d_out;
  float* avis = outq + (size_t)Bb * Nn * Cc;     // outputs concatenated in return order

  char* ws = (char*)d_ws;                        // ~16.03 MB used
  _Float16* qh = (_Float16*)(ws);                //  4 MB
  _Float16* kh = (_Float16*)(ws + (4u << 20));   //  4 MB
  _Float16* vT = (_Float16*)(ws + (8u << 20));   //  4 MB  [B,C,N] transposed
  _Float16* xb = (_Float16*)(ws + (12u << 20));  //  4 MB
  float* invm  = (float*)(ws + (16u << 20));     // 32 KB

  rowmax_inv_kernel<<<32, 256, 0, stream>>>(amask, invm);
  qkv_kernel<<<3072, 256, 0, stream>>>(query, Wq, bq, Wk, bk, Wv, bv, qh, kh, vT);
  attn_kernel<<<512, 256, 0, stream>>>(qh, kh, vT, amask, invm, avis, xb);
  outproj_kernel<<<1024, 256, 0, stream>>>(xb, Wo, bo, query, outq);
}
